// encoderLayer_15152644621124
// MI455X (gfx1250) — matrix operand order repaired and checked by reading, not compiled, not run on hardware
//
#include <hip/hip_runtime.h>
#include <hip/hip_bf16.h>
#include <stdint.h>

typedef __attribute__((ext_vector_type(16))) _Float16 v16h;
typedef __attribute__((ext_vector_type(8)))  float    v8f;
typedef __attribute__((ext_vector_type(4)))  unsigned int tdm_v4u;
typedef __attribute__((ext_vector_type(8)))  int          tdm_v8i;
typedef __attribute__((ext_vector_type(4)))  int          tdm_v4i;

#define B_   4
#define S_   1024
#define D_   1024
#define H_   16
#define DK_  64
#define LN_EPS 1e-5f

// ---------------------------------------------------------------------------
// WMMA fragment loaders (CDNA5 16x16x32 f16 layouts, CDNA5 ISA)
// A (16x32 f16): lane L, row M = L&15; lanes<16 hold K {0..7,16..23},
// lanes>=16 hold K {8..15,24..31} -> two contiguous 16B chunks per lane.
__device__ __forceinline__ v16h load_a_frag(const _Float16* base, int ld, int lane) {
  int r   = lane & 15;
  int seg = (lane >> 4) << 3;            // 0 or 8
  const _Float16* p = base + r * ld + seg;
  union { v16h h; uint4 q[2]; } u;
  u.q[0] = *(const uint4*)(p);
  u.q[1] = *(const uint4*)(p + 16);
  return u.h;
}
// B (32x16 f16): lane L, col N = L&15; lanes<16 hold K {0..7,16..23}, lanes>=16 K {8..15,24..31}.
// Expects B stored transposed (N x K row-major) with row stride ld.
__device__ __forceinline__ v16h load_b_frag(const _Float16* base, int ld, int lane) {
  int n   = lane & 15;
  int seg = (lane >> 4) << 3;            // 0 or 8
  const _Float16* p = base + n * ld + seg;
  union { v16h h; uint4 q[2]; } u;
  u.q[0] = *(const uint4*)(p);
  u.q[1] = *(const uint4*)(p + 16);
  return u.h;
}

__device__ __forceinline__ v8f wmma_f16(v16h a, v16h b, v8f c) {
  return __builtin_amdgcn_wmma_f32_16x16x32_f16(false, a, false, b, (short)0, c, false, false);
}

// ---------------------------------------------------------------------------
// Tensor Data Mover: 2D tile load (rows x 32 f16 elems, packed into LDS).
// D# layout per CDNA5 ISA (groups 2/3 zero => 2D).
// amdgpu-toolchain (clang-23) builtin form: 6 args
//   (uint32x4 g0, int32x8 g1, int32x4, int32x4, int32x8, i32 cpol)
__device__ __forceinline__ uint32_t lds_off(const void* p) {
  // generic pointer to LDS: low 32 bits are the LDS byte offset (aperture map)
  return (uint32_t)(uintptr_t)p;
}

__device__ __forceinline__ void tdm_load_tile_f16(uint32_t lds, const _Float16* g,
                                                  uint32_t rowStrideElems) {
  uint64_t ga = (uint64_t)(uintptr_t)g;
  tdm_v4u g0 = {
      1u,                                               // count=1, user mode
      lds,                                              // lds_addr (bytes)
      (uint32_t)ga,                                     // global_addr[31:0]
      (uint32_t)((ga >> 32) & 0x1FFFFFFu) | (2u << 30)  // global_addr[56:32] | type=2
  };
  const uint32_t dim0 = 0x40000000u;                    // huge tensor dims: no OOB clip
  const uint32_t dim1 = 0x40000000u;
  tdm_v8i g1;
  g1[0] = (int)(1u << 16);                              // wg_mask=0, data_size=1 (2B)
  g1[1] = (int)((dim0 & 0xFFFFu) << 16);                // tensor_dim0[15:0]
  g1[2] = (int)(((dim0 >> 16) & 0xFFFFu) |              // tensor_dim0[31:16]
                ((dim1 & 0xFFFFu) << 16));              // tensor_dim1[15:0]
  g1[3] = (int)(((dim1 >> 16) & 0xFFFFu) |              // tensor_dim1[31:16]
                (32u << 16));                           // tile_dim0 = 32 elems
  g1[4] = 128;                                          // tile_dim1 = 128 rows, tile_dim2=0
  g1[5] = (int)rowStrideElems;                          // tensor_dim0_stride[31:0]
  g1[6] = 0;                                            // stride0[47:32] | stride1[15:0]
  g1[7] = 0;
  tdm_v4i z4 = {0, 0, 0, 0};
  tdm_v8i z8 = {0, 0, 0, 0, 0, 0, 0, 0};
  __builtin_amdgcn_tensor_load_to_lds(g0, g1, z4, z4, z8, 0);
}

// ---------------------------------------------------------------------------
// block-wide (256 thr, 8 waves) sum reduction
__device__ __forceinline__ float block_sum256(float v, float* sbuf) {
#pragma unroll
  for (int m = 16; m >= 1; m >>= 1) v += __shfl_xor(v, m, 32);
  int wave = threadIdx.x >> 5;
  int lane = threadIdx.x & 31;
  if (lane == 0) sbuf[wave] = v;
  __syncthreads();
  if (wave == 0) {
    float t = (lane < 8) ? sbuf[lane] : 0.0f;
#pragma unroll
    for (int m = 4; m >= 1; m >>= 1) t += __shfl_xor(t, m, 32);
    if (lane == 0) sbuf[0] = t;
  }
  __syncthreads();
  float r = sbuf[0];
  __syncthreads();
  return r;
}

// ---------------------------------------------------------------------------
// mask[row] = any(|x[row,:]| > 0) ; lengths[b] = sum of mask rows (prefix-valid)
__global__ __launch_bounds__(256) void row_mask_kernel(const float* __restrict__ x,
                                                       float* __restrict__ mask) {
  __shared__ float sbuf[8];
  const int row = blockIdx.x;
  float s = 0.0f;
  for (int c = threadIdx.x; c < D_; c += 256) s += fabsf(x[(size_t)row * D_ + c]);
  float tot = block_sum256(s, sbuf);
  if (threadIdx.x == 0) mask[row] = (tot > 0.0f) ? 1.0f : 0.0f;
}

__global__ __launch_bounds__(256) void batch_len_kernel(const float* __restrict__ mask,
                                                        int* __restrict__ lengths) {
  __shared__ float sbuf[8];
  const int b = blockIdx.x;
  float s = 0.0f;
  for (int c = threadIdx.x; c < S_; c += 256) s += mask[b * S_ + c];
  float tot = block_sum256(s, sbuf);
  if (threadIdx.x == 0) lengths[b] = (int)(tot + 0.5f);
}

// ---------------------------------------------------------------------------
__global__ __launch_bounds__(256) void cvt_f16_kernel(const float* __restrict__ in,
                                                      _Float16* __restrict__ out, int n) {
  int i = blockIdx.x * 1024 + threadIdx.x;
#pragma unroll
  for (int k = 0; k < 4; ++k) {
    int idx = i + k * 256;
    if (idx < n) out[idx] = (_Float16)in[idx];
  }
}

// Wq (H,D,DK) f32 -> BT[(h*DK+e)][d] f16  (N x K, row-major, K = D)
__global__ __launch_bounds__(256) void pack_qkv_kernel(const float* __restrict__ W,
                                                       _Float16* __restrict__ BT) {
  int o = blockIdx.x * 256 + threadIdx.x;       // over D_*D_ outputs
  int d   = o % D_;
  int col = o / D_;
  int h   = col / DK_;
  int e   = col % DK_;
  BT[o] = (_Float16)W[((size_t)h * D_ + d) * DK_ + e];
}

// ---------------------------------------------------------------------------
// Tiled WMMA GEMM: C(MxN) = A(MxK f16 rowmajor) * BT(NxK f16 rowmajor)^T
// Block 256 thr = 8 waves (4x2), tile 128x128, BK=32.
// Staging: Tensor Data Mover, double-buffered; wave 0 issues 2 descriptors
// per stage (A,B) and tracks them with TENSORcnt; workgroup barrier publishes.
// EPI: 0 = store f16 ; 1 = relu(c+bias)*mask -> f16 ; 2 = (c+bias)*mask -> f32
template <int EPI>
__global__ __launch_bounds__(256) void gemm_wmma_kernel(
    const _Float16* __restrict__ A, const _Float16* __restrict__ BT,
    float* __restrict__ Cf, _Float16* __restrict__ Ch,
    const float* __restrict__ bias, const float* __restrict__ rowmask,
    int M, int N, int K) {
  __shared__ _Float16 As[2][128 * 32];
  __shared__ _Float16 Bs[2][128 * 32];
  const int tid  = threadIdx.x;
  const int lane = tid & 31;
  const int wave = tid >> 5;
  const int wm   = wave >> 1;          // 0..3
  const int wn   = wave & 1;           // 0..1
  const int m0   = blockIdx.y * 128;
  const int n0   = blockIdx.x * 128;

  const v8f vzero = {};
  v8f acc[2][4];
#pragma unroll
  for (int i = 0; i < 2; ++i)
#pragma unroll
    for (int j = 0; j < 4; ++j) acc[i][j] = vzero;

  const _Float16* Abase = A  + (size_t)m0 * K;
  const _Float16* Bbase = BT + (size_t)n0 * K;

  const int nstages = K >> 5;
  if (wave == 0) {
    tdm_load_tile_f16(lds_off(&As[0][0]), Abase, (uint32_t)K);
    tdm_load_tile_f16(lds_off(&Bs[0][0]), Bbase, (uint32_t)K);
  }

  for (int s = 0; s < nstages; ++s) {
    const int buf = s & 1;
    if (wave == 0) {
      if (s + 1 < nstages) {
        const int k1 = (s + 1) << 5;
        tdm_load_tile_f16(lds_off(&As[buf ^ 1][0]), Abase + k1, (uint32_t)K);
        tdm_load_tile_f16(lds_off(&Bs[buf ^ 1][0]), Bbase + k1, (uint32_t)K);
        __builtin_amdgcn_s_wait_tensorcnt(2);   // stage s landed; s+1 in flight
      } else {
        __builtin_amdgcn_s_wait_tensorcnt(0);   // final stage landed
      }
    }
    __syncthreads();

    v16h af[2], bf[4];
#pragma unroll
    for (int mt = 0; mt < 2; ++mt)
      af[mt] = load_a_frag(&As[buf][(wm * 32 + mt * 16) * 32], 32, lane);
#pragma unroll
    for (int nt = 0; nt < 4; ++nt)
      bf[nt] = load_b_frag(&Bs[buf][(wn * 64 + nt * 16) * 32], 32, lane);
#pragma unroll
    for (int mt = 0; mt < 2; ++mt)
#pragma unroll
      for (int nt = 0; nt < 4; ++nt)
        acc[mt][nt] = wmma_f16(af[mt], bf[nt], acc[mt][nt]);
    __syncthreads();   // all waves done reading buf before TDM refills it
  }

  const int cl = lane & 15;
  const int rh = (lane >> 4) << 3;
#pragma unroll
  for (int mt = 0; mt < 2; ++mt) {
#pragma unroll
    for (int nt = 0; nt < 4; ++nt) {
#pragma unroll
      for (int r = 0; r < 8; ++r) {
        int gm = m0 + wm * 32 + mt * 16 + rh + r;
        int gn = n0 + wn * 64 + nt * 16 + cl;
        float v = acc[mt][nt][r];
        size_t o = (size_t)gm * N + gn;
        if (EPI == 0) {
          Ch[o] = (_Float16)v;
        } else if (EPI == 1) {
          v = fmaxf(v + bias[gn], 0.0f) * rowmask[gm];
          Ch[o] = (_Float16)v;
        } else {
          v = (v + bias[gn]) * rowmask[gm];
          Cf[o] = v;
        }
      }
    }
  }
}

// ---------------------------------------------------------------------------
// Flash attention per (b, h, 128 q-rows). 8 waves x 16 q-rows each.
// K/V^T staged in LDS per 64-key block; online softmax in f32 registers;
// P re-entered through per-wave LDS scratch as f16 A-fragments.
__global__ __launch_bounds__(256) void flash_attn_kernel(
    const _Float16* __restrict__ Qh, const _Float16* __restrict__ Kh,
    const _Float16* __restrict__ Vh, const float* __restrict__ rowmask,
    const int* __restrict__ lengths, float* __restrict__ Z) {
  __shared__ _Float16 Kt[64 * 64];         // [key][e]
  __shared__ _Float16 VT[64 * 64];         // [e][key]
  __shared__ _Float16 Pb[8 * 16 * 64];     // per-wave P scratch

  const int b     = blockIdx.z;
  const int h     = blockIdx.y;
  const int qBase = blockIdx.x * 128;
  const int tid   = threadIdx.x;
  const int lane  = tid & 31;
  const int wave  = tid >> 5;
  const int len   = lengths[b];
  const int hOff  = h * DK_;
  const int ld    = H_ * DK_;              // 1024

  const _Float16* Qbase = Qh + ((size_t)(b * S_ + qBase + wave * 16)) * ld + hOff;
  v16h qf0 = load_a_frag(Qbase,      ld, lane);   // e 0..31
  v16h qf1 = load_a_frag(Qbase + 32, ld, lane);   // e 32..63

  const v8f vzero = {};
  v8f O[4];
  float mrow[8], lrow[8];
#pragma unroll
  for (int i = 0; i < 4; ++i) O[i] = vzero;
#pragma unroll
  for (int r = 0; r < 8; ++r) { mrow[r] = -__builtin_inff(); lrow[r] = 0.0f; }

  const int sr = tid >> 2;                 // 0..63
  const int sc = (tid & 3) * 16;           // 0,16,32,48
  const int cl = lane & 15;
  const int rh = (lane >> 4) << 3;
  _Float16* P = &Pb[wave * 16 * 64];

  const int nkb = (len + 63) >> 6;
  for (int kb = 0; kb < nkb; ++kb) {
    const size_t krow = (size_t)(b * S_ + kb * 64 + sr);
    const uint4* gK = (const uint4*)(Kh + krow * ld + hOff + sc);
    *(uint4*)&Kt[sr * 64 + sc]     = gK[0];
    *(uint4*)&Kt[sr * 64 + sc + 8] = gK[1];
    union { uint4 q[2]; _Float16 e[16]; } uv;
    const uint4* gV = (const uint4*)(Vh + krow * ld + hOff + sc);
    uv.q[0] = gV[0];
    uv.q[1] = gV[1];
#pragma unroll
    for (int i = 0; i < 16; ++i) VT[(sc + i) * 64 + sr] = uv.e[i];
    __syncthreads();

    // energy tile: S = Q (16x64) * K^T (64x64) -> 4 x (16x16)
    v8f sacc[4];
#pragma unroll
    for (int nt = 0; nt < 4; ++nt) {
      sacc[nt] = vzero;
      v16h b0 = load_b_frag(&Kt[(nt * 16) * 64],      64, lane);
      v16h b1 = load_b_frag(&Kt[(nt * 16) * 64 + 32], 64, lane);
      sacc[nt] = wmma_f16(qf0, b0, sacc[nt]);
      sacc[nt] = wmma_f16(qf1, b1, sacc[nt]);
    }

    // scale + column mask (cols >= len -> -inf)
    const float scale = 0.125f;              // 1/sqrt(DK)
#pragma unroll
    for (int nt = 0; nt < 4; ++nt) {
      int col = kb * 64 + nt * 16 + cl;
      bool bad = (col >= len);
#pragma unroll
      for (int r = 0; r < 8; ++r) {
        float e = sacc[nt][r] * scale;
        sacc[nt][r] = bad ? -__builtin_inff() : e;
      }
    }

    // online softmax (row stats replicated across 16 lanes of each half)
#pragma unroll
    for (int r = 0; r < 8; ++r) {
      float v = fmaxf(fmaxf(sacc[0][r], sacc[1][r]), fmaxf(sacc[2][r], sacc[3][r]));
      v = fmaxf(v, __shfl_xor(v, 1, 32));
      v = fmaxf(v, __shfl_xor(v, 2, 32));
      v = fmaxf(v, __shfl_xor(v, 4, 32));
      v = fmaxf(v, __shfl_xor(v, 8, 32));
      float mn = fmaxf(mrow[r], v);
      float alpha = __expf(mrow[r] - mn);
      float s = 0.0f;
#pragma unroll
      for (int nt = 0; nt < 4; ++nt) {
        float p = __expf(sacc[nt][r] - mn);
        sacc[nt][r] = p;
        s += p;
      }
      s += __shfl_xor(s, 1, 32);
      s += __shfl_xor(s, 2, 32);
      s += __shfl_xor(s, 4, 32);
      s += __shfl_xor(s, 8, 32);
      lrow[r] = lrow[r] * alpha + s;
      mrow[r] = mn;
#pragma unroll
      for (int et = 0; et < 4; ++et) O[et][r] *= alpha;
    }

    // P (C-layout) -> LDS -> A-fragment layout, then O += P * V
#pragma unroll
    for (int nt = 0; nt < 4; ++nt)
#pragma unroll
      for (int r = 0; r < 8; ++r)
        P[(rh + r) * 64 + nt * 16 + cl] = (_Float16)sacc[nt][r];

    v16h pf0 = load_a_frag(P,      64, lane);   // keys 0..31
    v16h pf1 = load_a_frag(P + 32, 64, lane);   // keys 32..63
#pragma unroll
    for (int et = 0; et < 4; ++et) {
      v16h vb0 = load_b_frag(&VT[(et * 16) * 64],      64, lane);
      v16h vb1 = load_b_frag(&VT[(et * 16) * 64 + 32], 64, lane);
      O[et] = wmma_f16(pf0, vb0, O[et]);
      O[et] = wmma_f16(pf1, vb1, O[et]);
    }
    __syncthreads();
  }

  // epilogue: O / l * m4 -> Z[b,s,h*64+e]
#pragma unroll
  for (int r = 0; r < 8; ++r) {
    int q = qBase + wave * 16 + rh + r;
    float inv = (lrow[r] > 0.0f) ? (1.0f / lrow[r]) : 0.0f;
    float mk = rowmask[b * S_ + q] * inv;
#pragma unroll
    for (int et = 0; et < 4; ++et)
      Z[((size_t)(b * S_ + q)) * ld + hOff + et * 16 + cl] = O[et][r] * mk;
  }
}

// ---------------------------------------------------------------------------
// out = LayerNorm(X + Y) * g + b ; optional f16 copy for next GEMM
__global__ __launch_bounds__(256) void add_ln_kernel(
    const float* __restrict__ X, const float* __restrict__ Y,
    const float* __restrict__ g, const float* __restrict__ bb,
    float* __restrict__ outF, _Float16* __restrict__ outH) {
  __shared__ float sbuf[8];
  const int row = blockIdx.x;
  const int t = threadIdx.x;
  const size_t base = (size_t)row * D_;
  float v[4];
  float s = 0.0f;
#pragma unroll
  for (int i = 0; i < 4; ++i) {
    int c = t + i * 256;
    v[i] = X[base + c] + Y[base + c];
    s += v[i];
  }
  float mu = block_sum256(s, sbuf) * (1.0f / D_);
  float s2 = 0.0f;
#pragma unroll
  for (int i = 0; i < 4; ++i) { float d = v[i] - mu; s2 += d * d; }
  float var = block_sum256(s2, sbuf) * (1.0f / D_);
  float rstd = rsqrtf(var + LN_EPS);
#pragma unroll
  for (int i = 0; i < 4; ++i) {
    int c = t + i * 256;
    float o = (v[i] - mu) * rstd * g[c] + bb[c];
    outF[base + c] = o;
    if (outH) outH[base + c] = (_Float16)o;
  }
}

// ---------------------------------------------------------------------------
extern "C" void kernel_launch(void* const* d_in, const int* in_sizes, int n_in,
                              void* d_out, int out_size, void* d_ws, size_t ws_size,
                              hipStream_t stream) {
  (void)in_sizes; (void)n_in; (void)out_size; (void)ws_size;
  const float* x     = (const float*)d_in[0];
  const float* Wq    = (const float*)d_in[1];
  const float* Wk    = (const float*)d_in[2];
  const float* Wv    = (const float*)d_in[3];
  const float* ln1_g = (const float*)d_in[4];
  const float* ln1_b = (const float*)d_in[5];
  const float* fc1_w = (const float*)d_in[6];
  const float* fc1_b = (const float*)d_in[7];
  const float* fc2_w = (const float*)d_in[8];
  const float* fc2_b = (const float*)d_in[9];
  const float* ln2_g = (const float*)d_in[10];
  const float* ln2_b = (const float*)d_in[11];
  float* out = (float*)d_out;

  char* ws = (char*)d_ws;
  size_t off = 0;
  auto alloc = [&](size_t bytes) {
    char* p = ws + off;
    off += (bytes + 255) & ~(size_t)255;
    return p;
  };
  const size_t NROW = (size_t)B_ * S_;                 // 4096
  const size_t NELT = NROW * D_;                       // 4M
  float*    mask    = (float*)alloc(NROW * 4);
  int*      lengths = (int*)alloc(64);
  _Float16* xh      = (_Float16*)alloc(NELT * 2);
  _Float16* BTq     = (_Float16*)alloc((size_t)D_ * D_ * 2);
  _Float16* BTk     = (_Float16*)alloc((size_t)D_ * D_ * 2);
  _Float16* BTv     = (_Float16*)alloc((size_t)D_ * D_ * 2);
  _Float16* fc1h    = (_Float16*)alloc((size_t)D_ * D_ * 2);
  _Float16* fc2h    = (_Float16*)alloc((size_t)D_ * D_ * 2);
  _Float16* Qhb     = (_Float16*)alloc(NELT * 2);
  _Float16* Khb     = (_Float16*)alloc(NELT * 2);
  _Float16* Vhb     = (_Float16*)alloc(NELT * 2);
  float*    Z       = (float*)alloc(NELT * 4);
  float*    hbuf    = (float*)alloc(NELT * 4);
  _Float16* hh      = Qhb;   // reuse: Q dead after attention
  _Float16* fbuf    = Khb;   // reuse: K dead after attention
  float*    f2      = Z;     // reuse: Z dead after first LN

  // mask + lengths
  row_mask_kernel<<<(unsigned)NROW, 256, 0, stream>>>(x, mask);
  batch_len_kernel<<<B_, 256, 0, stream>>>(mask, lengths);

  // f16 packs
  cvt_f16_kernel<<<(unsigned)(NELT / 1024), 256, 0, stream>>>(x, xh, (int)NELT);
  pack_qkv_kernel<<<D_ * D_ / 256, 256, 0, stream>>>(Wq, BTq);
  pack_qkv_kernel<<<D_ * D_ / 256, 256, 0, stream>>>(Wk, BTk);
  pack_qkv_kernel<<<D_ * D_ / 256, 256, 0, stream>>>(Wv, BTv);
  cvt_f16_kernel<<<D_ * D_ / 1024, 256, 0, stream>>>(fc1_w, fc1h, D_ * D_); // BT = fc1_w (B = fc1_w^T)
  cvt_f16_kernel<<<D_ * D_ / 1024, 256, 0, stream>>>(fc2_w, fc2h, D_ * D_);

  // QKV projections
  dim3 gG(D_ / 128, (unsigned)(NROW / 128));           // (8, 32)
  gemm_wmma_kernel<0><<<gG, 256, 0, stream>>>(xh, BTq, nullptr, Qhb, nullptr, nullptr, (int)NROW, D_, D_);
  gemm_wmma_kernel<0><<<gG, 256, 0, stream>>>(xh, BTk, nullptr, Khb, nullptr, nullptr, (int)NROW, D_, D_);
  gemm_wmma_kernel<0><<<gG, 256, 0, stream>>>(xh, BTv, nullptr, Vhb, nullptr, nullptr, (int)NROW, D_, D_);

  // flash attention -> Z (already (b,s,h,e) = reshape of transposed attn)
  dim3 gF(S_ / 128, H_, B_);
  flash_attn_kernel<<<gF, 256, 0, stream>>>(Qhb, Khb, Vhb, mask, lengths, Z);

  // h = LN(x + z)
  add_ln_kernel<<<(unsigned)NROW, 256, 0, stream>>>(x, Z, ln1_g, ln1_b, hbuf, hh);

  // FFN
  gemm_wmma_kernel<1><<<gG, 256, 0, stream>>>(hh, fc1h, nullptr, fbuf, fc1_b, mask, (int)NROW, D_, D_);
  gemm_wmma_kernel<2><<<gG, 256, 0, stream>>>(fbuf, fc2h, f2, nullptr, fc2_b, mask, (int)NROW, D_, D_);

  // out = LN(h + f)
  add_ln_kernel<<<(unsigned)NROW, 256, 0, stream>>>(hbuf, f2, ln2_g, ln2_b, out, nullptr);
}
